// KVCache_AttentionHead_41042707481096
// MI455X (gfx1250) — matrix operand order repaired and checked by reading, not compiled, not run on hardware
//
#include <hip/hip_runtime.h>
#include <hip/hip_bf16.h>
#include <stdint.h>

// ---------------------------------------------------------------------------
// Flash attention (causal + dropout), B=4, L=4096, H=64, gfx1250.
//  - bf16 WMMA (v_wmma_f32_16x16x32_bf16) for QK^T and PV, fp32 softmax state
//  - double-buffered async global->LDS staging (ASYNCcnt) for K/V tiles
//  - scalarized uniform control flow, batched dropout RNG (4 draws/hash)
// ---------------------------------------------------------------------------

typedef __attribute__((ext_vector_type(16))) __bf16 v16bf;
typedef __attribute__((ext_vector_type(8)))  float  v8f;

#define B_SZ    4
#define L_SZ    4096
#define H_SZ    64
#define QT      64     // q rows per workgroup (4 waves x 16 rows)
#define KT      32     // key tile (Bc)
#define PITCHK  72     // bf16 pitch for Ks rows (144 B, 16B aligned)
#define PITCHV  40     // bf16 pitch for Vs/Ps rows (80 B, 16B aligned)

#define LDS_AS __attribute__((address_space(3)))

static __device__ inline __bf16 f2bf(float f) { return (__bf16)f; }

struct Pack32 { uint4 a, b; };

// Build a v16bf A/B operand from two 16-byte-aligned 8xbf16 runs in LDS.
static __device__ inline v16bf ld2_v16bf(const __bf16* p0, const __bf16* p1) {
  Pack32 p;
  p.a = *(const uint4*)p0;
  p.b = *(const uint4*)p1;
  return __builtin_bit_cast(v16bf, p);
}

// Async DMA: 16 bytes/lane, global -> LDS, tracked by ASYNCcnt.
static __device__ inline void async_ld_b128(const float* g, void* l) {
  unsigned loff = (unsigned)(unsigned long long)(LDS_AS char*)(char*)l;
  unsigned long long ga = (unsigned long long)g;
  asm volatile("global_load_async_to_lds_b128 %0, %1, off"
               :: "v"(loff), "v"(ga) : "memory");
}

static __device__ inline void wait_async0() {
  asm volatile("s_wait_asynccnt 0x0" ::: "memory");
}

static __device__ inline unsigned hash3(unsigned a, unsigned b, unsigned c) {
  unsigned h = a * 0x9E3779B1u ^ b * 0x85EBCA77u ^ c * 0xC2B2AE3Du;
  h ^= h >> 16; h *= 0x7FEB352Du; h ^= h >> 15; h *= 0x846CA68Bu; h ^= h >> 16;
  return h;
}

__global__ __launch_bounds__(128)
void fa_causal_dropout_kernel(const float* __restrict__ q,
                              const float* __restrict__ k,
                              const float* __restrict__ v,
                              float* __restrict__ out) {
  __shared__ float  KsF[2][KT][H_SZ];      // async staging, fp32, double-buffered
  __shared__ float  VsF[2][KT][H_SZ];
  __shared__ __bf16 Ks[KT][PITCHK];        // K tile bf16, row-major [key][h]
  __shared__ __bf16 Vs[H_SZ][PITCHV];      // V tile bf16, transposed [h][key]
  __shared__ __bf16 Ps[4][16][PITCHV];     // per-wave P staging [row][key]

  const int nqblk = L_SZ / QT;
  const int b     = blockIdx.x / nqblk;
  const int qblk  = blockIdx.x % nqblk;
  const int q0    = qblk * QT;

  const int tid  = threadIdx.x;
  const int w    = tid >> 5;
  const int lane = tid & 31;
  const int half = lane >> 4;   // 16-lane half
  const int row  = lane & 15;   // row (A) or column (B/C) index
  const int wq   = q0 + w * 16; // wave's q-row base
  // Provably wave-uniform copy -> scalar compares / s_cbranch instead of EXEC ops
  const int wq_s = __builtin_amdgcn_readfirstlane(wq);

  const float ksc      = 0.18033688011112042f; // (1/sqrt(64))*log2(e)
  const float keep_inv = 1.0f / 0.9f;

  // ---- Q tile into A-layout registers (16x64 bf16 = 2 chunks of 16x32) ----
  v16bf aq[2];
  {
    const float* qrow = q + ((size_t)b * L_SZ + wq + row) * H_SZ;
    for (int c = 0; c < 2; ++c)
      for (int g = 0; g < 2; ++g) {
        int hb = 32 * c + 16 * g + 8 * half;
        float4 f0 = *(const float4*)(qrow + hb);
        float4 f1 = *(const float4*)(qrow + hb + 4);
        int e = 8 * g;
        aq[c][e+0]=f2bf(f0.x); aq[c][e+1]=f2bf(f0.y);
        aq[c][e+2]=f2bf(f0.z); aq[c][e+3]=f2bf(f0.w);
        aq[c][e+4]=f2bf(f1.x); aq[c][e+5]=f2bf(f1.y);
        aq[c][e+6]=f2bf(f1.z); aq[c][e+7]=f2bf(f1.w);
      }
  }

  v8f o[4];
  for (int n = 0; n < 4; ++n)
    for (int r = 0; r < 8; ++r) o[n][r] = 0.0f;
  float mrow[8], lrow[8];
  for (int r = 0; r < 8; ++r) { mrow[r] = -3.0e38f; lrow[r] = 0.0f; }

  const int jmax = 2 * qblk + 1;          // causal: 32-key tiles 0..jmax

  // Issue async DMA of K/V tile jt into staging buffer jt&1.
  // 32x64 floats = 512 float4 per matrix; 128 threads x 4 each.
  auto issue_tile = [&](int jt) {
    const int kb  = jt * KT;
    const int buf = jt & 1;
    for (int i = 0; i < 4; ++i) {
      int idx = tid + i * 128;
      int r   = idx >> 4;           // key row 0..31
      int c4  = (idx & 15) * 4;     // h base
      const size_t goff = ((size_t)b * L_SZ + kb + r) * H_SZ + c4;
      async_ld_b128(k + goff, &KsF[buf][r][c4]);
      async_ld_b128(v + goff, &VsF[buf][r][c4]);
    }
  };

  issue_tile(0);

  for (int j = 0; j <= jmax; ++j) {
    const int k0 = j * KT;
    wait_async0();          // my wave's DMA for tile j done
    __syncthreads();        // everyone's DMA for tile j done

    if (j < jmax) issue_tile(j + 1);               // overlap with convert+compute
    if (j + 2 <= jmax) {                           // warm L2 for tile j+2
      const size_t poff = ((size_t)b * L_SZ + (j + 2) * KT) * H_SZ + (size_t)tid * 16;
      __builtin_prefetch(k + poff, 0, 1);
      __builtin_prefetch(v + poff, 0, 1);
    }

    // ---- convert staging fp32 -> bf16 operand tiles (V transposed) ----
    {
      const int buf = j & 1;
      for (int i = 0; i < 4; ++i) {
        int idx = tid + i * 128;
        int r   = idx >> 4;
        int c4  = (idx & 15) * 4;
        float4 kf = *(const float4*)&KsF[buf][r][c4];
        float4 vf = *(const float4*)&VsF[buf][r][c4];
        __bf16* kp = &Ks[r][c4];
        kp[0] = f2bf(kf.x); kp[1] = f2bf(kf.y);
        kp[2] = f2bf(kf.z); kp[3] = f2bf(kf.w);
        Vs[c4+0][r] = f2bf(vf.x); Vs[c4+1][r] = f2bf(vf.y);
        Vs[c4+2][r] = f2bf(vf.z); Vs[c4+3][r] = f2bf(vf.w);
      }
    }
    __syncthreads();        // Ks/Vs ready

    if (k0 <= wq_s + 15) {  // scalar-uniform: wave has unmasked keys in tile
      // ---- S = Q K^T : 2 tiles of 16x16, each contraction K=64 -> 2 WMMAs
      v8f s[2];
      for (int n = 0; n < 2; ++n) {
        v8f acc;
        for (int r = 0; r < 8; ++r) acc[r] = 0.0f;
        for (int c = 0; c < 2; ++c) {
          const __bf16* bp = &Ks[16 * n + row][32 * c + 8 * half];
          v16bf kb = ld2_v16bf(bp, bp + 16);
          acc = __builtin_amdgcn_wmma_f32_16x16x32_bf16(
                    false, aq[c], false, kb, (short)0, acc, false, false);
        }
        s[n] = acc;
      }

      // ---- causal mask, per 16-key sub-tile (uniform skip below diagonal)
      for (int n = 0; n < 2; ++n) {
        if (k0 + 16 * n + 15 > wq_s) {
          int key = k0 + 16 * n + row;
          for (int r = 0; r < 8; ++r) {
            int qr = wq + r + 8 * half;
            if (key > qr) s[n][r] = -3.0e38f;
          }
        }
      }

      // ---- online softmax ----
      float pmax[8];
      for (int r = 0; r < 8; ++r) {
        float m = fmaxf(s[0][r], s[1][r]);
        for (int off = 8; off; off >>= 1) m = fmaxf(m, __shfl_xor(m, off));
        pmax[r] = m;
      }
      float rs[8];
      for (int r = 0; r < 8; ++r) {
        float mn   = fmaxf(mrow[r], pmax[r]);
        float corr = exp2f((mrow[r] - mn) * ksc);
        mrow[r] = mn;
        lrow[r] *= corr;
        for (int n = 0; n < 4; ++n) o[n][r] *= corr;
        rs[r] = 0.0f;
      }

      // p = exp2((s-m)*ksc); ell from pre-dropout p; dropout -> Ps (bf16).
      // RNG: one 32-bit hash supplies four 8-bit Bernoulli draws (keep: <230).
      for (int n = 0; n < 2; ++n) {
        int key = k0 + 16 * n + row;
        unsigned h0 = hash3((unsigned)b, (unsigned)key, (unsigned)(wq + 8 * half));
        unsigned h1 = hash3((unsigned)b, (unsigned)key, (unsigned)(wq + 8 * half + 4));
        for (int r = 0; r < 8; ++r) {
          float p = exp2f((s[n][r] - mrow[r]) * ksc);
          rs[r] += p;
          unsigned byte = ((r < 4 ? h0 : h1) >> (8 * (r & 3))) & 0xFFu;
          float pd = (byte < 230u) ? p * keep_inv : 0.0f;
          Ps[w][r + 8 * half][16 * n + row] = f2bf(pd);
        }
      }
      for (int r = 0; r < 8; ++r) {
        float t = rs[r];
        for (int off = 8; off; off >>= 1) t += __shfl_xor(t, off);
        lrow[r] += t;
      }

      // ---- O += P V : reload P in A-layout (within-wave LDS round-trip) ----
      v16bf pa;
      {
        const __bf16* p0 = &Ps[w][row][8 * half];
        pa = ld2_v16bf(p0, p0 + 16);
      }
      for (int nh = 0; nh < 4; ++nh) {
        const __bf16* bp = &Vs[16 * nh + row][8 * half];
        v16bf vb = ld2_v16bf(bp, bp + 16);
        o[nh] = __builtin_amdgcn_wmma_f32_16x16x32_bf16(
                    false, pa, false, vb, (short)0, o[nh], false, false);
      }
    }
    // next iteration's top barrier separates this compute from the next
    // convert pass that overwrites Ks/Vs
  }

  // ---- epilogue: O / ell, fp32 store ----
  for (int r = 0; r < 8; ++r) {
    float inv = 1.0f / lrow[r];
    int qr = wq + r + 8 * half;
    float* orow = out + ((size_t)b * L_SZ + qr) * H_SZ;
    for (int nh = 0; nh < 4; ++nh)
      orow[16 * nh + row] = o[nh][r] * inv;
  }
}

extern "C" void kernel_launch(void* const* d_in, const int* in_sizes, int n_in,
                              void* d_out, int out_size, void* d_ws, size_t ws_size,
                              hipStream_t stream) {
  (void)in_sizes; (void)n_in; (void)out_size; (void)d_ws; (void)ws_size;
  const float* q = (const float*)d_in[0];
  const float* k = (const float*)d_in[1];
  const float* v = (const float*)d_in[2];
  float* out = (float*)d_out;
  dim3 grid(B_SZ * (L_SZ / QT));
  dim3 block(128);
  fa_causal_dropout_kernel<<<grid, block, 0, stream>>>(q, k, v, out);
}
